// PhaGatModel2_38405597561072
// MI455X (gfx1250) — matrix operand order repaired and checked by reading, not compiled, not run on hardware
//
#include <hip/hip_runtime.h>

#define N_NODES 32768
#define NSEG    (N_NODES + 1)
#define N_EDGES 262144
#define EMB     64
#define B_MOL   2048
#define MAX_PER_MOL 32

typedef __attribute__((ext_vector_type(8)))  __bf16 v8bf;
typedef __attribute__((ext_vector_type(16))) __bf16 v16bf;
typedef __attribute__((ext_vector_type(8)))  float  v8f;

__device__ __forceinline__ unsigned short f2bf(float f) {
  unsigned int u = __float_as_uint(f);
  unsigned int r = (u + 0x7FFFu + ((u >> 16) & 1u)) >> 16;  // round-to-nearest-even
  return (unsigned short)r;
}

__device__ __forceinline__ float wave_sum(float v) {
#pragma unroll
  for (int o = 16; o > 0; o >>= 1) v += __shfl_down(v, o, 32);
  return v;
}

// ---------------- embeddings ----------------

// h_orig[(N_NODES+1) x 64]; row 0 = zeros, row 1+i = node_feats[i] @ W_emb + b_emb
__global__ void k_embed_nodes(const float* __restrict__ nf, const float* __restrict__ W,
                              const float* __restrict__ b, float* __restrict__ h_orig) {
  int idx = blockIdx.x * blockDim.x + threadIdx.x;
  if (idx >= NSEG * EMB) return;
  int node = idx >> 6, c = idx & 63;
  float acc = 0.0f;
  if (node > 0) {
    acc = b[c];
    const float* row = nf + (size_t)(node - 1) * 9;
#pragma unroll
    for (int f = 0; f < 9; ++f) acc += row[f] * W[f * EMB + c];
  }
  h_orig[idx] = acc;
}

// message0 (bf16, E x 64) = [fdist | rij] @ W_dist + b_dist
__global__ void k_embed_msg(const float* __restrict__ fdist, const float* __restrict__ rij,
                            const float* __restrict__ W, const float* __restrict__ b,
                            unsigned short* __restrict__ msg) {
  int idx = blockIdx.x * blockDim.x + threadIdx.x;
  if (idx >= N_EDGES * EMB) return;
  int e = idx >> 6, c = idx & 63;
  float acc = b[c];
  const float* row = fdist + (size_t)e * 9;
#pragma unroll
  for (int f = 0; f < 9; ++f) acc += row[f] * W[f * EMB + c];
  acc += rij[e] * W[9 * EMB + c];
  msg[idx] = f2bf(acc);
}

// Wt[n][k] = bf16(W[k][n]) so the WMMA B operand loads contiguously per lane
__global__ void k_transpose_w(const float* __restrict__ W, unsigned short* __restrict__ Wt, int d) {
  int idx = blockIdx.x * blockDim.x + threadIdx.x;
  if (idx >= d * d) return;
  int n = idx / d, k = idx - n * d;
  Wt[idx] = f2bf(W[(size_t)k * d + n]);
}

// ---------------- WMMA GEMM: H(MxK,f32) = A(MxK,bf16) @ W(KxK) + bias ----------------
// Register-blocked: one wave owns 1 M-tile x NTB N-tiles; one A fragment feeds NTB WMMAs.
// K is compile-time so the K-loop fully unrolls into immediate-offset b128 loads.
template <int K, int NTB>
__global__ void k_gemm_bf16_t(const unsigned short* __restrict__ A,
                              const unsigned short* __restrict__ Bt,
                              const float* __restrict__ bias,
                              float* __restrict__ H, int M) {
  constexpr int N = K;  // square layers
  const int wave = (int)((blockIdx.x * blockDim.x + threadIdx.x) >> 5);
  const int lane = (int)(threadIdx.x & 31);
  const int mtiles = M >> 4;
  constexpr int ngroups = (N / 16) / NTB;
  if (wave >= mtiles * ngroups) return;
  const int ng = wave / mtiles;          // adjacent waves share the same B tiles (L0/L2 reuse)
  const int mt = wave - ng * mtiles;
  const int idx16 = lane & 15;
  const int hi = lane >> 4;
  // A layout (contiguous k = 16*hi + s, same order as B): lane holds row idx16, contiguous 16-elem k-chunk at hi*16
  const unsigned short* arow = A + (size_t)(mt * 16 + idx16) * K + (hi << 4);
  // B layout (contiguous k = 16*hi + s, same order as A): lane holds column idx16, contiguous 16-elem k-chunk at hi*16
  const int n0 = ng * NTB * 16;
  const unsigned short* brow = Bt + (size_t)(n0 + idx16) * K + (hi << 4);

  v8f acc[NTB];
#pragma unroll
  for (int j = 0; j < NTB; ++j) acc[j] = (v8f){};

#pragma unroll
  for (int k = 0; k < K; k += 32) {
    v8bf alo = *reinterpret_cast<const v8bf*>(arow + k);
    v8bf ahi = *reinterpret_cast<const v8bf*>(arow + k + 8);
    v16bf a;
#pragma unroll
    for (int i = 0; i < 8; ++i) { a[i] = alo[i]; a[i + 8] = ahi[i]; }
#pragma unroll
    for (int j = 0; j < NTB; ++j) {
      v16bf bm = *reinterpret_cast<const v16bf*>(brow + (size_t)j * 16 * K + k);
      acc[j] = __builtin_amdgcn_wmma_f32_16x16x32_bf16(false, a, false, bm, (short)0, acc[j],
                                                       false, false);
    }
  }

  // D layout: VGPR r, lane -> row (r + 8*hi), col idx16
  const int mbase = mt * 16 + (hi << 3);
#pragma unroll
  for (int j = 0; j < NTB; ++j) {
    const int n = n0 + j * 16 + idx16;
    const float bv = bias[n];
    float* hout = H + (size_t)mbase * N + n;
#pragma unroll
    for (int r = 0; r < 8; ++r) hout[(size_t)r * N] = acc[j][r] + bv;
  }
}

// ---------------- attention ----------------

// nd0[s] = h_orig[s] . a[0:64], nd1[s] = h_orig[s] . a[64:128]; one wave per node
__global__ void k_node_dots(const float* __restrict__ h_orig, const float* __restrict__ a,
                            float* __restrict__ nd0, float* __restrict__ nd1) {
  int wave = (int)((blockIdx.x * blockDim.x + threadIdx.x) >> 5);
  int lane = (int)(threadIdx.x & 31);
  if (wave >= NSEG) return;
  const float* hrow = h_orig + (size_t)wave * EMB;
  float v0 = hrow[lane] * a[lane] + hrow[lane + 32] * a[lane + 32];
  float v1 = hrow[lane] * a[64 + lane] + hrow[lane + 32] * a[96 + lane];
  v0 = wave_sum(v0);
  v1 = wave_sum(v1);
  if (lane == 0) { nd0[wave] = v0; nd1[wave] = v1; }
}

__global__ void k_init_attn(unsigned int* __restrict__ segmax, float* __restrict__ den,
                            float* __restrict__ num, int d) {
  int idx = blockIdx.x * blockDim.x + threadIdx.x;
  int total = NSEG * d;
  if (idx >= total) return;
  if (idx < NSEG) { segmax[idx] = 0u; den[idx] = 0.0f; }
  num[idx] = 0.0f;
}

// logits[e] = leaky_relu(nd0[s0] + nd1[s1] + h[e].a[128:128+d]); atomic segment max
__global__ void k_logits(const float* __restrict__ h, const float* __restrict__ a,
                         const int* __restrict__ see, const int* __restrict__ bscope,
                         const float* __restrict__ nd0, const float* __restrict__ nd1,
                         float* __restrict__ logits, unsigned int* __restrict__ segmax, int d) {
  int wave = (int)((blockIdx.x * blockDim.x + threadIdx.x) >> 5);
  int lane = (int)(threadIdx.x & 31);
  if (wave >= N_EDGES) return;
  const float* hrow = h + (size_t)wave * d;
  float acc = 0.0f;
  for (int c = lane; c < d; c += 32) acc += hrow[c] * a[128 + c];
  acc = wave_sum(acc);
  if (lane == 0) {
    int s0 = see[wave * 2 + 0];
    int s1 = see[wave * 2 + 1];
    float x = nd0[s0] + nd1[s1] + acc;
    x = x > 0.0f ? x : 0.2f * x;
    logits[wave] = x;
    unsigned int u = __float_as_uint(x);
    unsigned int key = (u & 0x80000000u) ? ~u : (u | 0x80000000u);  // order-preserving encode
    atomicMax(&segmax[bscope[wave]], key);
  }
}

__global__ void k_expsum(const float* __restrict__ logits, const unsigned int* __restrict__ segmax,
                         const int* __restrict__ bscope, float* __restrict__ ex,
                         float* __restrict__ den) {
  int e = blockIdx.x * blockDim.x + threadIdx.x;
  if (e >= N_EDGES) return;
  int s = bscope[e];
  unsigned int key = segmax[s];
  unsigned int u = (key & 0x80000000u) ? (key & 0x7FFFFFFFu) : ~key;  // decode
  float m = __uint_as_float(u);
  float v = __expf(logits[e] - m);
  ex[e] = v;
  atomicAdd(&den[s], v);
}

// num[seg][c] += ex[e] * h[e][c]  (native global_atomic_add_f32)
__global__ void k_scatter_num(const float* __restrict__ h, const float* __restrict__ ex,
                              const int* __restrict__ bscope, float* __restrict__ num, int d) {
  long long idx = (long long)blockIdx.x * blockDim.x + threadIdx.x;
  long long total = (long long)N_EDGES * d;
  if (idx >= total) return;
  int e = (int)(idx / d);
  int c = (int)(idx - (long long)e * d);
  atomicAdd(&num[(size_t)bscope[e] * d + c], ex[e] * h[(size_t)e * d + c]);
}

// tf[s] = [ elu(num/den) (d cols) | h_orig[s] (64 cols) ]
__global__ void k_finalize_tf(const float* __restrict__ num, const float* __restrict__ den,
                              const float* __restrict__ h_orig, float* __restrict__ tf, int d) {
  int w = d + EMB;
  long long idx = (long long)blockIdx.x * blockDim.x + threadIdx.x;
  long long total = (long long)NSEG * w;
  if (idx >= total) return;
  int s = (int)(idx / w);
  int c = (int)(idx - (long long)s * w);
  float v;
  if (c < d) {
    float dn = den[s];
    float x = dn > 0.0f ? num[(size_t)s * d + c] / dn : 0.0f;
    v = x > 0.0f ? x : (__expf(x) - 1.0f);  // ELU
  } else {
    v = h_orig[(size_t)s * EMB + (c - d)];
  }
  tf[(size_t)s * w + c] = v;
}

// message = bf16( 0.5 * (tf[lig] + tf[upd]) ), width w
__global__ void k_msg_update(const float* __restrict__ tf, const int* __restrict__ lig,
                             const int* __restrict__ upd, unsigned short* __restrict__ msg, int w) {
  long long idx = (long long)blockIdx.x * blockDim.x + threadIdx.x;
  long long total = (long long)N_EDGES * w;
  if (idx >= total) return;
  int e = (int)(idx / w);
  int c = (int)(idx - (long long)e * w);
  float m = 0.5f * (tf[(size_t)lig[e] * w + c] + tf[(size_t)upd[e] * w + c]);
  msg[(size_t)e * w + c] = f2bf(m);
}

// out[b][c] = sum_j tf[l_scope[b][j]][c], tf width 256
__global__ void k_output(const float* __restrict__ tf, const int* __restrict__ lscope,
                         float* __restrict__ out) {
  int idx = blockIdx.x * blockDim.x + threadIdx.x;
  if (idx >= B_MOL * 256) return;
  int b = idx >> 8, c = idx & 255;
  const int* row = lscope + b * MAX_PER_MOL;
  float acc = 0.0f;
#pragma unroll 4
  for (int j = 0; j < MAX_PER_MOL; ++j) acc += tf[(size_t)row[j] * 256 + c];
  out[idx] = acc;
}

// ---------------- host launcher ----------------

static inline int nblk(long long n) { return (int)((n + 255) / 256); }

extern "C" void kernel_launch(void* const* d_in, const int* in_sizes, int n_in,
                              void* d_out, int out_size, void* d_ws, size_t ws_size,
                              hipStream_t stream) {
  const float* node_feats = (const float*)d_in[0];
  const float* fdist      = (const float*)d_in[1];
  const float* rij        = (const float*)d_in[2];
  const int*   see        = (const int*)d_in[3];
  const int*   bscope     = (const int*)d_in[4];
  const int*   upd        = (const int*)d_in[5];
  const int*   lig        = (const int*)d_in[6];
  const int*   lscope     = (const int*)d_in[7];
  const float* W_emb      = (const float*)d_in[8];
  const float* b_emb      = (const float*)d_in[9];
  const float* W_dist     = (const float*)d_in[10];
  const float* b_dist     = (const float*)d_in[11];
  const float* fcW[3]  = {(const float*)d_in[12], (const float*)d_in[15], (const float*)d_in[18]};
  const float* fcb[3]  = {(const float*)d_in[13], (const float*)d_in[16], (const float*)d_in[19]};
  const float* attn[3] = {(const float*)d_in[14], (const float*)d_in[17], (const float*)d_in[20]};

  size_t off = 0;
  char* base = (char*)d_ws;
  auto wsalloc = [&](size_t bytes) -> void* {
    off = (off + 255) & ~(size_t)255;
    void* p = base + off;
    off += bytes;
    return p;
  };

  float*          h_orig = (float*)wsalloc((size_t)NSEG * EMB * sizeof(float));
  float*          tf     = (float*)wsalloc((size_t)NSEG * 256 * sizeof(float));
  unsigned short* msg    = (unsigned short*)wsalloc((size_t)N_EDGES * 192 * sizeof(unsigned short));
  float*          hbuf   = (float*)wsalloc((size_t)N_EDGES * 192 * sizeof(float));
  float*          logits = (float*)wsalloc((size_t)N_EDGES * sizeof(float));
  float*          ex     = (float*)wsalloc((size_t)N_EDGES * sizeof(float));
  float*          nd0    = (float*)wsalloc((size_t)NSEG * sizeof(float));
  float*          nd1    = (float*)wsalloc((size_t)NSEG * sizeof(float));
  unsigned int*   segmax = (unsigned int*)wsalloc((size_t)NSEG * sizeof(unsigned int));
  float*          den    = (float*)wsalloc((size_t)NSEG * sizeof(float));
  float*          num    = (float*)wsalloc((size_t)NSEG * 192 * sizeof(float));
  unsigned short* Wt     = (unsigned short*)wsalloc((size_t)192 * 192 * sizeof(unsigned short));

  // Embeddings
  k_embed_nodes<<<nblk((long long)NSEG * EMB), 256, 0, stream>>>(node_feats, W_emb, b_emb, h_orig);
  k_embed_msg<<<nblk((long long)N_EDGES * EMB), 256, 0, stream>>>(fdist, rij, W_dist, b_dist, msg);

  const int dims[3] = {64, 128, 192};
  for (int i = 0; i < 3; ++i) {
    const int d = dims[i];
    // Transposed bf16 weights for WMMA B operand
    k_transpose_w<<<nblk((long long)d * d), 256, 0, stream>>>(fcW[i], Wt, d);
    // h = msg @ fc_W + fc_b   (WMMA bf16 -> f32, 1x4 register blocking)
    {
      const int mtiles = N_EDGES / 16;
      const int ngroups = (d / 16) / 4;
      long long threads = (long long)mtiles * ngroups * 32;
      switch (d) {
        case 64:
          k_gemm_bf16_t<64, 4><<<nblk(threads), 256, 0, stream>>>(msg, Wt, fcb[i], hbuf, N_EDGES);
          break;
        case 128:
          k_gemm_bf16_t<128, 4><<<nblk(threads), 256, 0, stream>>>(msg, Wt, fcb[i], hbuf, N_EDGES);
          break;
        default:
          k_gemm_bf16_t<192, 4><<<nblk(threads), 256, 0, stream>>>(msg, Wt, fcb[i], hbuf, N_EDGES);
          break;
      }
    }
    // Attention logits + segment softmax
    k_node_dots<<<nblk((long long)NSEG * 32), 256, 0, stream>>>(h_orig, attn[i], nd0, nd1);
    k_init_attn<<<nblk((long long)NSEG * d), 256, 0, stream>>>(segmax, den, num, d);
    k_logits<<<nblk((long long)N_EDGES * 32), 256, 0, stream>>>(hbuf, attn[i], see, bscope,
                                                                nd0, nd1, logits, segmax, d);
    k_expsum<<<nblk((long long)N_EDGES), 256, 0, stream>>>(logits, segmax, bscope, ex, den);
    k_scatter_num<<<nblk((long long)N_EDGES * d), 256, 0, stream>>>(hbuf, ex, bscope, num, d);
    // tf = [ elu(softmax-weighted h) | h_orig ]
    k_finalize_tf<<<nblk((long long)NSEG * (d + EMB)), 256, 0, stream>>>(num, den, h_orig, tf, d);
    // message for next layer
    if (i < 2) {
      k_msg_update<<<nblk((long long)N_EDGES * (d + EMB)), 256, 0, stream>>>(tf, lig, upd, msg,
                                                                             d + EMB);
    }
  }

  // Final per-molecule gather-sum (tf width 256)
  k_output<<<nblk((long long)B_MOL * 256), 256, 0, stream>>>(tf, lscope, (float*)d_out);

  (void)in_sizes; (void)n_in; (void)out_size; (void)ws_size;
}
